// TrellisFusedDenseMLP_21182778703898
// MI455X (gfx1250) — matrix operand order repaired and checked by reading, not compiled, not run on hardware
//
#include <hip/hip_runtime.h>
#include <hip/hip_bf16.h>

// ---------------------------------------------------------------------------
// Types for CDNA5 WMMA (wave32)
// ---------------------------------------------------------------------------
typedef __attribute__((ext_vector_type(16))) __bf16 v16bf;
typedef __attribute__((ext_vector_type(8)))  __bf16 v8bf;
typedef __attribute__((ext_vector_type(8)))  float  v8f;

union FragU { v16bf v; v8bf h[2]; };

// A-matrix 16x32 (MxK), 16-bit layout (ISA 7.12.2):
//   lane L: M = L&15. L<16 -> elems0..7 = K0..7,  elems8..15 = K16..23
//           L>=16-> elems0..7 = K8..15, elems8..15 = K24..31
__device__ __forceinline__ v16bf load_frag_a(const __bf16* __restrict__ tile,
                                             int ldk, int lane) {
  const __bf16* p = tile + (size_t)(lane & 15) * ldk + ((lane & 16) ? 8 : 0);
  FragU f;
  f.h[0] = *(const v8bf*)(p);
  f.h[1] = *(const v8bf*)(p + 16);
  return f.v;
}

// B-matrix 32x16 (KxN) built from row-major W (N x K):
//   lane L: N = L&15. L<16 -> elems0..7 = K0..7, elems8..15 = K16..23
//           L>=16-> elems0..7 = K8..15, elems8..15 = K24..31 (same as A).
__device__ __forceinline__ v16bf load_frag_b(const __bf16* __restrict__ tile,
                                             int ldk, int lane) {
  const __bf16* p = tile + (size_t)(lane & 15) * ldk + ((lane & 16) ? 8 : 0);
  FragU f;
  f.h[0] = *(const v8bf*)(p);
  f.h[1] = *(const v8bf*)(p + 16);
  return f.v;
}

__device__ __forceinline__ v8f wmma_bf16(v16bf a, v16bf b, v8f c) {
  // (neg_a, A, neg_b, B, c_mod, C, reuse_a, reuse_b)
  return __builtin_amdgcn_wmma_f32_16x16x32_bf16(false, a, false, b,
                                                 (short)0, c, false, false);
}

// ---------------------------------------------------------------------------
// Kernel 0: f32 -> bf16 conversion of activations
// ---------------------------------------------------------------------------
__global__ __launch_bounds__(256) void k_f32_to_bf16(const float* __restrict__ in,
                                                     __bf16* __restrict__ out,
                                                     size_t n) {
  size_t i = ((size_t)blockIdx.x * blockDim.x + threadIdx.x) * 4;
  if (i + 3 < n) {
    float4 v = *(const float4*)(in + i);
    out[i + 0] = (__bf16)v.x;
    out[i + 1] = (__bf16)v.y;
    out[i + 2] = (__bf16)v.z;
    out[i + 3] = (__bf16)v.w;
  }
}

// ---------------------------------------------------------------------------
// Kernel 1/2: dequantize trellis weights to bf16
//   W[r,c] = grid16[idx[r,c]] * scales[r, c/128] * srow[r] * scol[c]
// ---------------------------------------------------------------------------
__global__ __launch_bounds__(256) void k_dequant(const int*   __restrict__ idx,
                                                 const float* __restrict__ sc,
                                                 const float* __restrict__ srow,
                                                 const float* __restrict__ scol,
                                                 const float* __restrict__ grid16,
                                                 __bf16* __restrict__ out,
                                                 int C) {
  const int r = blockIdx.y;
  const int c = blockIdx.x * 256 + threadIdx.x;
  const size_t off = (size_t)r * C + c;
  const float w = grid16[idx[off] & 15] *
                  sc[(size_t)r * (C >> 7) + (c >> 7)] *
                  srow[r] * scol[c];
  out[off] = (__bf16)w;
}

// ---------------------------------------------------------------------------
// Kernel 3: GEMM1 fused  inter = silu(x @ Wg^T) * (x @ Wu^T)   (bf16 out)
//   X: M x K (K=H), Wgu: (2I) x K row-major, inter: M x I
//   Block tile 64(M) x 128(N gate + matching up), 8 waves (2x4),
//   each wave: 32x32 gate + 32x32 up, K-step 32.
//   8 WMMA per 12 b128 loads per wave per K-step.
// ---------------------------------------------------------------------------
__global__ __launch_bounds__(256) void k_gemm1_silu(const __bf16* __restrict__ X,
                                                    const __bf16* __restrict__ Wgu,
                                                    __bf16* __restrict__ inter,
                                                    int M, int K, int I) {
  const int lane = threadIdx.x & 31;
  const int wave = threadIdx.x >> 5;
  const int wm = wave >> 2;   // 0..1
  const int wn = wave & 3;    // 0..3
  const int m0 = blockIdx.y * 64 + wm * 32;
  const int n0 = blockIdx.x * 128 + wn * 32;

  v8f ag[2][2] = {{{}, {}}, {{}, {}}};
  v8f au[2][2] = {{{}, {}}, {{}, {}}};

  const __bf16* a0  = X + (size_t)m0 * K;
  const __bf16* a1  = X + (size_t)(m0 + 16) * K;
  const __bf16* bg0 = Wgu + (size_t)n0 * K;             // gate rows
  const __bf16* bg1 = Wgu + (size_t)(n0 + 16) * K;
  const __bf16* bu0 = Wgu + (size_t)(n0 + I) * K;       // up rows
  const __bf16* bu1 = Wgu + (size_t)(n0 + I + 16) * K;

  for (int k = 0; k < K; k += 32) {
    v16bf A0  = load_frag_a(a0 + k, K, lane);
    v16bf A1  = load_frag_a(a1 + k, K, lane);
    v16bf Bg0 = load_frag_b(bg0 + k, K, lane);
    v16bf Bg1 = load_frag_b(bg1 + k, K, lane);
    v16bf Bu0 = load_frag_b(bu0 + k, K, lane);
    v16bf Bu1 = load_frag_b(bu1 + k, K, lane);
    ag[0][0] = wmma_bf16(A0, Bg0, ag[0][0]);
    ag[1][0] = wmma_bf16(A1, Bg0, ag[1][0]);
    ag[0][1] = wmma_bf16(A0, Bg1, ag[0][1]);
    ag[1][1] = wmma_bf16(A1, Bg1, ag[1][1]);
    au[0][0] = wmma_bf16(A0, Bu0, au[0][0]);
    au[1][0] = wmma_bf16(A1, Bu0, au[1][0]);
    au[0][1] = wmma_bf16(A0, Bu1, au[0][1]);
    au[1][1] = wmma_bf16(A1, Bu1, au[1][1]);
  }

  // C/D f32 16x16 layout: lane L, reg r -> M = r + (L<16?0:8), N = L&15
  const int coll = lane & 15;
  const int rb   = (lane & 16) ? 8 : 0;
#pragma unroll
  for (int mi = 0; mi < 2; ++mi) {
#pragma unroll
    for (int ni = 0; ni < 2; ++ni) {
      const int col = n0 + ni * 16 + coll;
#pragma unroll
      for (int r = 0; r < 8; ++r) {
        const int m = m0 + mi * 16 + rb + r;
        float g = ag[mi][ni][r], u = au[mi][ni][r];
        float s = (g / (1.0f + __expf(-g))) * u;
        inter[(size_t)m * I + col] = (__bf16)s;
      }
    }
  }
}

// ---------------------------------------------------------------------------
// Kernel 4: GEMM2  out = inter @ Wd^T   (f32 out)
//   inter: M x K (K=I), Wd: N x K row-major (N=H), out: M x N
//   Block tile 64x128, 8 waves (2x4), each wave 32x32.
// ---------------------------------------------------------------------------
__global__ __launch_bounds__(256) void k_gemm2(const __bf16* __restrict__ A,
                                               const __bf16* __restrict__ W,
                                               float* __restrict__ out,
                                               int M, int K, int N) {
  const int lane = threadIdx.x & 31;
  const int wave = threadIdx.x >> 5;
  const int wm = wave >> 2;   // 0..1
  const int wn = wave & 3;    // 0..3
  const int m0 = blockIdx.y * 64 + wm * 32;
  const int n0 = blockIdx.x * 128 + wn * 32;

  v8f c[2][2] = {{{}, {}}, {{}, {}}};

  const __bf16* a0 = A + (size_t)m0 * K;
  const __bf16* a1 = A + (size_t)(m0 + 16) * K;
  const __bf16* b0 = W + (size_t)n0 * K;
  const __bf16* b1 = W + (size_t)(n0 + 16) * K;

  for (int k = 0; k < K; k += 32) {
    v16bf A0 = load_frag_a(a0 + k, K, lane);
    v16bf A1 = load_frag_a(a1 + k, K, lane);
    v16bf B0 = load_frag_b(b0 + k, K, lane);
    v16bf B1 = load_frag_b(b1 + k, K, lane);
    c[0][0] = wmma_bf16(A0, B0, c[0][0]);
    c[1][0] = wmma_bf16(A1, B0, c[1][0]);
    c[0][1] = wmma_bf16(A0, B1, c[0][1]);
    c[1][1] = wmma_bf16(A1, B1, c[1][1]);
  }

  const int coll = lane & 15;
  const int rb   = (lane & 16) ? 8 : 0;
#pragma unroll
  for (int mi = 0; mi < 2; ++mi) {
#pragma unroll
    for (int ni = 0; ni < 2; ++ni) {
      const int col = n0 + ni * 16 + coll;
#pragma unroll
      for (int r = 0; r < 8; ++r) {
        out[(size_t)(m0 + mi * 16 + rb + r) * N + col] = c[mi][ni][r];
      }
    }
  }
}

// ---------------------------------------------------------------------------
// Launcher
// inputs (setup_inputs order):
//  0:x f32 (B,S,H)  1:gu_indices i32 (2I,H)  2:gu_scales f32 (2I,H/128)
//  3:gu_su f32 (H)  4:gu_sv f32 (2I)         5:d_indices i32 (H,I)
//  6:d_scales f32 (H,I/128)  7:d_su f32 (I)  8:d_sv f32 (H)
//  9:grid f32 (16)  10:intermediate_size (scalar)
// ---------------------------------------------------------------------------
extern "C" void kernel_launch(void* const* d_in, const int* in_sizes, int n_in,
                              void* d_out, int out_size, void* d_ws, size_t ws_size,
                              hipStream_t stream) {
  const float* x      = (const float*)d_in[0];
  const int*   gu_idx = (const int*)  d_in[1];
  const float* gu_sc  = (const float*)d_in[2];
  const float* gu_su  = (const float*)d_in[3];
  const float* gu_sv  = (const float*)d_in[4];
  const int*   dd_idx = (const int*)  d_in[5];
  const float* dd_sc  = (const float*)d_in[6];
  const float* dd_su  = (const float*)d_in[7];
  const float* dd_sv  = (const float*)d_in[8];
  const float* grid16 = (const float*)d_in[9];
  float* out = (float*)d_out;

  const int H    = in_sizes[3];            // 2048
  const int twoI = in_sizes[4];            // 11264
  const int I    = in_sizes[7];            // 5632
  const size_t M = (size_t)in_sizes[0] / (size_t)H;  // B*S = 8192

  // Workspace layout (all 16B-aligned; total ~186 MB)
  char* ws = (char*)d_ws;
  __bf16* xb    = (__bf16*)ws;                 size_t off = M * (size_t)H * 2;
  __bf16* wgu   = (__bf16*)(ws + off);         off += (size_t)twoI * H * 2;
  __bf16* wd    = (__bf16*)(ws + off);         off += (size_t)H * I * 2;
  __bf16* inter = (__bf16*)(ws + off);

  // 0) x -> bf16
  {
    size_t n = M * (size_t)H;
    k_f32_to_bf16<<<dim3((unsigned)(n / 1024)), 256, 0, stream>>>(x, xb, n);
  }
  // 1) dequant W_gu (2I x H): srow = gu_sv, scol = gu_su
  k_dequant<<<dim3(H / 256, twoI), 256, 0, stream>>>(gu_idx, gu_sc, gu_sv, gu_su,
                                                     grid16, wgu, H);
  // 2) dequant W_d (H x I): srow = d_sv, scol = d_su
  k_dequant<<<dim3(I / 256, H), 256, 0, stream>>>(dd_idx, dd_sc, dd_sv, dd_su,
                                                  grid16, wd, I);
  // 3) fused gate/up GEMM + SiLU -> inter (M x I, bf16)
  k_gemm1_silu<<<dim3(I / 128, (unsigned)(M / 64)), 256, 0, stream>>>(
      xb, wgu, inter, (int)M, H, I);
  // 4) down GEMM -> out (M x H, f32)
  k_gemm2<<<dim3(H / 128, (unsigned)(M / 64)), 256, 0, stream>>>(
      inter, wd, out, (int)M, I, H);
}
